// MOE_7782480740679
// MI455X (gfx1250) — matrix operand order repaired and checked by reading, not compiled, not run on hardware
//
#include <hip/hip_runtime.h>

typedef __bf16 bf16_t;
typedef __bf16 v16bf __attribute__((ext_vector_type(16)));
typedef __bf16 v8bf  __attribute__((ext_vector_type(8)));
typedef float  v8f   __attribute__((ext_vector_type(8)));
typedef int    moe_v4i __attribute__((vector_size(16)));

namespace moe {
constexpr int E = 8, CAP = 1280, D = 1024, H = 4096, NB = 4, NS = 2048;
constexpr int NTOK = NB * NS; // 8192
}

// CDNA5 async global->LDS path (ASYNCcnt-tracked), guarded so either toolchain compiles.
#if defined(__HIP_DEVICE_COMPILE__) && __has_builtin(__builtin_amdgcn_global_load_async_to_lds_b128)
#define MOE_ASYNC_LDS 1
#define MOE_GPTR(p) ((__attribute__((address_space(1))) moe_v4i*)(uintptr_t)(p))
#define MOE_LPTR(p) ((__attribute__((address_space(3))) moe_v4i*)(uint32_t)(uintptr_t)(p))
#if __has_builtin(__builtin_amdgcn_s_wait_asynccnt)
#define MOE_WAIT_ASYNC(n) __builtin_amdgcn_s_wait_asynccnt(n)
#else
#define MOE_WAIT_ASYNC(n) asm volatile("s_wait_asynccnt %0" ::"i"(n) : "memory")
#endif
#else
#define MOE_ASYNC_LDS 0
#endif

// ---------------- transpose + fp32 -> bf16 (in [Z][R][Cc] f32 -> out [Z][Cc][R] bf16) ---
__global__ void transpose_to_bf16(const float* __restrict__ in, bf16_t* __restrict__ out,
                                  int R, int Cc) {
  __shared__ float tile[32][33];
  int z = blockIdx.z;
  const float* ip = in + (long)z * R * Cc;
  bf16_t* op = out + (long)z * R * Cc;
  int c0 = blockIdx.x * 32, r0 = blockIdx.y * 32;
  int tx = threadIdx.x, ty = threadIdx.y;
  #pragma unroll
  for (int j = 0; j < 32; j += 8)
    tile[ty + j][tx] = ip[(long)(r0 + ty + j) * Cc + c0 + tx];
  __syncthreads();
  #pragma unroll
  for (int j = 0; j < 32; j += 8)
    op[(long)(c0 + ty + j) * R + r0 + tx] = (bf16_t)tile[tx][ty + j];
}

// ---------------- router: one wave per token, top-1 argmax (softmax is monotone) --------
__global__ void router_kernel(const float* __restrict__ x, const float* __restrict__ rw,
                              const float* __restrict__ rb, int* __restrict__ eid) {
  int wave = threadIdx.x >> 5;
  int lane = threadIdx.x & 31;
  int tok = blockIdx.x * 8 + wave;
  float acc[8] = {0.f,0.f,0.f,0.f,0.f,0.f,0.f,0.f};
  const float* xr = x + (long)tok * moe::D;
  for (int d = lane; d < moe::D; d += 32) {
    float xv = xr[d];
    const float4* w = (const float4*)(rw + (long)d * 8);
    float4 w0 = w[0], w1 = w[1];
    acc[0] += xv * w0.x; acc[1] += xv * w0.y; acc[2] += xv * w0.z; acc[3] += xv * w0.w;
    acc[4] += xv * w1.x; acc[5] += xv * w1.y; acc[6] += xv * w1.z; acc[7] += xv * w1.w;
  }
  #pragma unroll
  for (int e = 0; e < 8; ++e)
    #pragma unroll
    for (int off = 16; off > 0; off >>= 1)
      acc[e] += __shfl_xor(acc[e], off, 32);
  if (lane == 0) {
    float best = acc[0] + rb[0]; int be = 0;
    #pragma unroll
    for (int e = 1; e < 8; ++e) { float v = acc[e] + rb[e]; if (v > best) { best = v; be = e; } }
    eid[tok] = be;
  }
}

// ---------------- exact token-order rank per expert + overflow compaction ----------------
__global__ void rank_kernel(const int* __restrict__ eid, int* __restrict__ pos,
                            int* __restrict__ valid, int* __restrict__ ovidx,
                            int* __restrict__ meta) {
  __shared__ int ovcnt[8];
  __shared__ int base[8];
  int t = threadIdx.x;
  if (t < 8) {
    int cnt = 0;
    for (int i = 0; i < moe::NTOK; ++i) {
      if (eid[i] == t) {
        int r = cnt++;
        bool ok = r < moe::CAP;
        pos[i] = ok ? r : 0;
        valid[i] = ok ? 1 : 0;
      }
    }
    ovcnt[t] = (cnt > moe::CAP) ? (cnt - moe::CAP) : 0;
  }
  __syncthreads();
  if (t == 0) {
    int s = 0;
    for (int e = 0; e < 8; ++e) { base[e] = s; s += ovcnt[e]; }
    meta[0] = s;                    // overflow token count
    meta[1] = (s + 127) & ~127;     // padded to GEMM M-tile
  }
  __syncthreads();
  if (t < 8) {
    int j = base[t];
    for (int i = 0; i < moe::NTOK; ++i)
      if (eid[i] == t && !valid[i]) ovidx[j++] = i;
  }
}

// ---------------- helpers ----------------------------------------------------------------
__global__ void zero16(uint4* __restrict__ p, long n) {
  long i = (long)blockIdx.x * blockDim.x + threadIdx.x;
  if (i < n) p[i] = make_uint4(0u, 0u, 0u, 0u);
}

__global__ void scatter_kernel(const float* __restrict__ x, const int* __restrict__ eid,
                               const int* __restrict__ pos, const int* __restrict__ valid,
                               bf16_t* __restrict__ buf) {
  int n = blockIdx.x;
  if (!valid[n]) return;
  long dst = ((long)eid[n] * moe::CAP + pos[n]) * moe::D;
  const float* src = x + (long)n * moe::D;
  for (int d = threadIdx.x; d < moe::D; d += blockDim.x) buf[dst + d] = (bf16_t)src[d];
}

__global__ void build_ovx(const float* __restrict__ x, const int* __restrict__ ovidx,
                          const int* __restrict__ meta, bf16_t* __restrict__ ovx) {
  int j = blockIdx.x;
  if (j >= meta[1]) return;
  bf16_t* dst = ovx + (long)j * moe::D;
  if (j < meta[0]) {
    const float* src = x + (long)ovidx[j] * moe::D;
    for (int d = threadIdx.x; d < moe::D; d += blockDim.x) dst[d] = (bf16_t)src[d];
  } else {
    for (int d = threadIdx.x; d < moe::D; d += blockDim.x) dst[d] = (bf16_t)0.0f;
  }
}

__global__ void combine_kernel(const float* __restrict__ y, const int* __restrict__ eid,
                               const int* __restrict__ pos, const int* __restrict__ valid,
                               float* __restrict__ out) {
  int n = blockIdx.x;
  if (!valid[n]) return;
  const float* src = y + ((long)eid[n] * moe::CAP + pos[n]) * moe::D;
  float* dst = out + (long)n * moe::D;
  for (int d = threadIdx.x; d < moe::D; d += blockDim.x) dst[d] = src[d];
}

// ---------------- WMMA GEMM: C[M,N] = act(A[M,K] * Bt[N,K]^T + bias) ---------------------
// A, Bt bf16 row-major; 128x128 block tile, BK=64, 8 waves of 32x64, 2x4 wmma accumulators.
// LDS rows padded to LDA=72 bf16 (36 dwords): gcd(36,64)=4 -> 16 rows/half-wave cover all
// 64 banks with disjoint 4-bank ranges => conflict-free ds_load_b128 fragment reads.
// Staging: CDNA5 global_load_async_to_lds_b128 with double-buffered LDS when available,
// else register-staged software pipeline (loads for tile k+1 issued before compute of k).
template <bool RELU, bool OUTBF16, bool SCATTER, bool DYNM>
__global__ void __launch_bounds__(256)
gemm_bt(const bf16_t* __restrict__ A, const bf16_t* __restrict__ Bt,
        const float* __restrict__ bias, void* __restrict__ Cout,
        int M, int N, int K,
        long sAz, long sBz, int sBiasZ, long sCz,
        const int* __restrict__ row_map, const int* __restrict__ meta) {
  constexpr int BM = 128, BN = 128, BK = 64;
  constexpr int LDA = BK + 8;  // padded LDS row stride (bf16 elems), 144B: 16B-aligned
#if MOE_ASYNC_LDS
  __shared__ bf16_t sA[2][BM * LDA];
  __shared__ bf16_t sB[2][BN * LDA];
#else
  __shared__ bf16_t sA[1][BM * LDA];
  __shared__ bf16_t sB[1][BN * LDA];
#endif

  const int m0 = blockIdx.y * BM;
  if constexpr (DYNM) { if (m0 >= meta[1]) return; }
  const int n0 = blockIdx.x * BN;
  const int z  = blockIdx.z;

  const int tid  = threadIdx.x;
  const int lane = tid & 31;
  const int wave = tid >> 5;
  const int wm = wave & 3;   // 4 waves over M (32 rows each)
  const int wn = wave >> 2;  // 2 waves over N (64 cols each)

  const bf16_t* Ab = A + (long)z * sAz + (long)m0 * K;
  const bf16_t* Bb = Bt + (long)z * sBz + (long)n0 * K;

  v8f acc[2][4] = {};

  const int srow = tid >> 3;        // 0..31
  const int scol = (tid & 7) * 8;   // 0,8,..,56

  auto compute = [&](const bf16_t* pA, const bf16_t* pB) {
    #pragma unroll
    for (int ks = 0; ks < BK; ks += 32) {
      union { v16bf v; v8bf h[2]; } af[2], bq[4];
      // A frag 16x32: lanes 0-15 -> K 0..7 & 16..23, lanes 16-31 -> K 8..15 & 24..31
      const int h8 = (lane >> 4) * 8;
      #pragma unroll
      for (int mt = 0; mt < 2; ++mt) {
        int r = wm * 32 + mt * 16 + (lane & 15);
        af[mt].h[0] = *(const v8bf*)&pA[r * LDA + ks + h8];
        af[mt].h[1] = *(const v8bf*)&pA[r * LDA + ks + h8 + 16];
      }
      // B frag 32x16: lanes 0-15 -> K 0..7 & 16..23, lanes 16-31 -> K 8..15 & 24..31 (column per lane)
      const int h16 = (lane >> 4) * 8;
      #pragma unroll
      for (int nt = 0; nt < 4; ++nt) {
        int c = wn * 64 + nt * 16 + (lane & 15);
        bq[nt].h[0] = *(const v8bf*)&pB[c * LDA + ks + h16];
        bq[nt].h[1] = *(const v8bf*)&pB[c * LDA + ks + h16 + 16];
      }
      #pragma unroll
      for (int mt = 0; mt < 2; ++mt)
        #pragma unroll
        for (int nt = 0; nt < 4; ++nt)
          acc[mt][nt] = __builtin_amdgcn_wmma_f32_16x16x32_bf16(
              false, af[mt].v, false, bq[nt].v, (short)0, acc[mt][nt], false, false);
    }
  };

#if MOE_ASYNC_LDS
  auto stage = [&](int b, int k0) {
    #pragma unroll
    for (int p = 0; p < 4; ++p) {
      int r = srow + p * 32;
      __builtin_amdgcn_global_load_async_to_lds_b128(
          MOE_GPTR(&Ab[(long)r * K + k0 + scol]),
          MOE_LPTR(&sA[b][r * LDA + scol]), 0, 0);
      __builtin_amdgcn_global_load_async_to_lds_b128(
          MOE_GPTR(&Bb[(long)r * K + k0 + scol]),
          MOE_LPTR(&sB[b][r * LDA + scol]), 0, 0);
    }
  };
  stage(0, 0);
  for (int k0 = 0; k0 < K; k0 += BK) {
    const int cur = (k0 >> 6) & 1;
    if (k0 + BK < K) {
      stage(cur ^ 1, k0 + BK);   // prefetch next tile into other half
      MOE_WAIT_ASYNC(8);         // current tile's 8 copies done; next 8 in flight
    } else {
      MOE_WAIT_ASYNC(0);
    }
    __syncthreads();             // all waves' async copies visible
    compute(sA[cur], sB[cur]);
    __syncthreads();             // reads done before next iteration overwrites other half
  }
#else
  uint4 ra[4], rb4[4];
  auto gload = [&](int k0) {
    #pragma unroll
    for (int p = 0; p < 4; ++p) {
      int r = srow + p * 32;
      ra[p]  = *(const uint4*)&Ab[(long)r * K + k0 + scol];
      rb4[p] = *(const uint4*)&Bb[(long)r * K + k0 + scol];
    }
  };
  gload(0);
  for (int k0 = 0; k0 < K; k0 += BK) {
    __syncthreads();             // previous compute finished reading LDS
    #pragma unroll
    for (int p = 0; p < 4; ++p) {
      int r = srow + p * 32;
      *(uint4*)&sA[0][r * LDA + scol] = ra[p];
      *(uint4*)&sB[0][r * LDA + scol] = rb4[p];
    }
    __syncthreads();             // LDS tile ready
    if (k0 + BK < K) gload(k0 + BK);  // issue next tile's loads under the WMMAs
    compute(sA[0], sB[0]);
  }
#endif

  // epilogue: C/D layout -> VGPR r holds M = (lane>=16)*8 + r, N = lane&15
  int mlim = 0;
  if constexpr (SCATTER) mlim = meta[0];
  #pragma unroll
  for (int nt = 0; nt < 4; ++nt) {
    const int col = n0 + wn * 64 + nt * 16 + (lane & 15);
    const float bv = bias[(long)z * sBiasZ + col];
    #pragma unroll
    for (int mt = 0; mt < 2; ++mt) {
      #pragma unroll
      for (int r = 0; r < 8; ++r) {
        int m = m0 + wm * 32 + mt * 16 + (lane >> 4) * 8 + r;
        float v = acc[mt][nt][r] + bv;
        if (RELU) v = fmaxf(v, 0.0f);
        long idx;
        if constexpr (SCATTER) {
          if (m >= mlim) continue;
          idx = (long)row_map[m] * N + col;
        } else {
          idx = (long)z * sCz + (long)m * N + col;
        }
        if constexpr (OUTBF16) ((bf16_t*)Cout)[idx] = (bf16_t)v;
        else                   ((float*)Cout)[idx]  = v;
      }
    }
  }
}

// ---------------- host orchestration -----------------------------------------------------
extern "C" void kernel_launch(void* const* d_in, const int* in_sizes, int n_in,
                              void* d_out, int out_size, void* d_ws, size_t ws_size,
                              hipStream_t stream) {
  using namespace moe;
  (void)in_sizes; (void)n_in; (void)out_size; (void)ws_size;
  const float* x   = (const float*)d_in[0];
  const float* rw  = (const float*)d_in[1];
  const float* rb  = (const float*)d_in[2];
  const float* w1  = (const float*)d_in[3];
  const float* b1  = (const float*)d_in[4];
  const float* w2  = (const float*)d_in[5];
  const float* b2  = (const float*)d_in[6];
  const float* ow1 = (const float*)d_in[7];
  const float* ob1 = (const float*)d_in[8];
  const float* ow2 = (const float*)d_in[9];
  const float* ob2 = (const float*)d_in[10];
  float* out = (float*)d_out;

  char* ws = (char*)d_ws;
  auto take = [&](long bytes) { char* p = ws; ws += (bytes + 255) & ~255L; return p; };
  bf16_t* w1t  = (bf16_t*)take(2L * E * H * D);     // [E][H][D]
  bf16_t* w2t  = (bf16_t*)take(2L * E * D * H);     // [E][D][H]
  bf16_t* ow1t = (bf16_t*)take(2L * H * D);         // [H][D]
  bf16_t* ow2t = (bf16_t*)take(2L * D * H);         // [D][H]
  bf16_t* buf  = (bf16_t*)take(2L * E * CAP * D);   // dispatched tokens
  bf16_t* hbuf = (bf16_t*)take(2L * E * CAP * H);   // hidden
  float*  ybuf = (float*) take(4L * E * CAP * D);   // expert outputs
  bf16_t* ovx  = (bf16_t*)take(2L * NTOK * D);      // overflow inputs (compacted)
  bf16_t* hov  = (bf16_t*)take(2L * NTOK * H);      // overflow hidden
  int* eid   = (int*)take(4L * NTOK);
  int* pos   = (int*)take(4L * NTOK);
  int* valid = (int*)take(4L * NTOK);
  int* ovidx = (int*)take(4L * NTOK);
  int* meta  = (int*)take(64);

  // weights -> bf16, pre-transposed so GEMM B-operand is [N][K] row-major
  transpose_to_bf16<<<dim3(H / 32, D / 32, E), dim3(32, 8), 0, stream>>>(w1, w1t, D, H);
  transpose_to_bf16<<<dim3(D / 32, H / 32, E), dim3(32, 8), 0, stream>>>(w2, w2t, H, D);
  transpose_to_bf16<<<dim3(H / 32, D / 32, 1), dim3(32, 8), 0, stream>>>(ow1, ow1t, D, H);
  transpose_to_bf16<<<dim3(D / 32, H / 32, 1), dim3(32, 8), 0, stream>>>(ow2, ow2t, H, D);

  router_kernel<<<NTOK / 8, 256, 0, stream>>>(x, rw, rb, eid);
  rank_kernel<<<1, 32, 0, stream>>>(eid, pos, valid, ovidx, meta);

  long bufBytes = 2L * E * CAP * D;
  zero16<<<(int)((bufBytes / 16 + 255) / 256), 256, 0, stream>>>((uint4*)buf, bufBytes / 16);
  scatter_kernel<<<NTOK, 256, 0, stream>>>(x, eid, pos, valid, buf);
  build_ovx<<<NTOK, 256, 0, stream>>>(x, ovidx, meta, ovx);

  // expert FFN: h = relu(buf @ w1 + b1), y = h @ w2 + b2
  gemm_bt<true, true, false, false><<<dim3(H / 128, CAP / 128, E), 256, 0, stream>>>(
      buf, w1t, b1, hbuf, CAP, H, D,
      (long)CAP * D, (long)H * D, H, (long)CAP * H, nullptr, nullptr);
  gemm_bt<false, false, false, false><<<dim3(D / 128, CAP / 128, E), 256, 0, stream>>>(
      hbuf, w2t, b2, ybuf, CAP, D, H,
      (long)CAP * H, (long)D * H, D, (long)CAP * D, nullptr, nullptr);

  // overflow path (dynamic M, usually zero tiles survive the meta[1] guard)
  gemm_bt<true, true, false, true><<<dim3(H / 128, NTOK / 128, 1), 256, 0, stream>>>(
      ovx, ow1t, ob1, hov, NTOK, H, D, 0, 0, 0, 0, nullptr, meta);
  gemm_bt<false, false, true, true><<<dim3(D / 128, NTOK / 128, 1), 256, 0, stream>>>(
      hov, ow2t, ob2, out, NTOK, D, H, 0, 0, 0, 0, ovidx, meta);

  combine_kernel<<<NTOK, 256, 0, stream>>>(ybuf, eid, pos, valid, out);
}
